// SegMinkUNet_63513976373528
// MI455X (gfx1250) — matrix operand order repaired and checked by reading, not compiled, not run on hardware
//
#include <hip/hip_runtime.h>

// MinkUNet-style dense 3D U-Net on MI455X (gfx1250), wave32 + WMMA.
// Activations channels-last [D,H,W,C] f16; f32 WMMA accumulate.
// Conv = implicit GEMM: each wave computes a 32(voxel) x 16(channel) tile,
// K = Cin per tap via v_wmma_f32_16x16x32_f16; B fragments staged in LDS.
// CIN/KS are template parameters so all K-chunk edge handling folds away.

typedef _Float16 h16;
typedef __attribute__((ext_vector_type(16))) _Float16 v16h;
typedef __attribute__((ext_vector_type(8)))  _Float16 v8h;
typedef __attribute__((ext_vector_type(8)))  float    v8f;
typedef __attribute__((ext_vector_type(4)))  unsigned int v4u;
typedef __attribute__((ext_vector_type(2)))  unsigned int v2u;

#define BN_INV 0.9999950000374997f   // 1/sqrt(1+1e-5)
#define GRID 64
#define INV_VSIZE 20.0f

// ---------------- voxelize ----------------
__global__ __launch_bounds__(256) void vox_scatter_kernel(
    const float* __restrict__ feats, const float* __restrict__ coords,
    float* __restrict__ gsum, float* __restrict__ gcnt, int N)
{
  int p = blockIdx.x * 256 + threadIdx.x;
  if (p >= N) return;
  int i0 = min(max((int)floorf(coords[p*3+0] * INV_VSIZE), 0), GRID-1);
  int i1 = min(max((int)floorf(coords[p*3+1] * INV_VSIZE), 0), GRID-1);
  int i2 = min(max((int)floorf(coords[p*3+2] * INV_VSIZE), 0), GRID-1);
  int v = (i0 * GRID + i1) * GRID + i2;
  #pragma unroll
  for (int c = 0; c < 4; ++c)
    atomicAdd(&gsum[v*4 + c], feats[p*4 + c]);
  atomicAdd(&gcnt[v], 1.0f);
}

__global__ __launch_bounds__(256) void vox_final_kernel(
    const float* __restrict__ gsum, const float* __restrict__ gcnt,
    h16* __restrict__ g16, int V)
{
  int v = blockIdx.x * 256 + threadIdx.x;
  if (v >= V) return;
  float c = fmaxf(gcnt[v], 1.0f);
  #pragma unroll
  for (int ch = 0; ch < 4; ++ch)
    g16[v*4 + ch] = (h16)(gsum[v*4 + ch] / c);
}

// ---------------- weight re-tile into per-lane WMMA B fragments ----------------
// dst[(((nt*taps + tap)*kch + kc)*32 + L)*16 + e] =
//     W[co = nt*16 + (L&15)][ci = kc*32 + (e>>3)*16 + (L>>4)*8 + (e&7)][tap' (mirrored for tconv)]
__global__ __launch_bounds__(256) void retile_w_kernel(
    const float* __restrict__ w, h16* __restrict__ dst,
    int Cin, int Cout, int taps, int kch, int ntiles, int mirror, int total)
{
  int i = blockIdx.x * 256 + threadIdx.x;
  if (i >= total) return;
  int e = i & 15;
  int L = (i >> 4) & 31;
  int rest = i >> 9;
  int kc = rest % kch; rest /= kch;
  int tap = rest % taps;
  int nt = rest / taps;
  int ci = kc*32 + (e >> 3)*16 + (L >> 4)*8 + (e & 7);
  int co = nt*16 + (L & 15);
  int st = mirror ? (taps - 1 - tap) : tap;
  float v = (ci < Cin && co < Cout) ? w[((long)co * Cin + ci) * taps + st] : 0.0f;
  dst[i] = (h16)v;
}

// Branchless A fragment load; all group decisions fold at compile time.
template<int CIN>
__device__ __forceinline__ v16h load_afrag(
    const h16* __restrict__ in, long rb, bool valid, int kc, int khalf)
{
  v16h a;
  if constexpr (CIN == 4) {
    // only K=0..3 nonzero; they live in elements 0..3 of lanes with khalf==0
    v2u u = *(const v2u*)(in + rb);             // 8B load, aligned (rb = vox*4)
    v2u z = {};
    u = (valid & (khalf == 0)) ? u : z;
    union { v2u u; h16 h[4]; } cv; cv.u = u;
    #pragma unroll
    for (int j = 0; j < 4; ++j) a[j] = cv.h[j];
    #pragma unroll
    for (int j = 4; j < 16; ++j) a[j] = (h16)0.0f;
    return a;
  }
  #pragma unroll
  for (int g = 0; g < 2; ++g) {
    int cbase = kc*32 + g*16;                   // constant after unroll
    if (cbase + 16 <= CIN) {                    // compile-time fold
      v4u u = *(const v4u*)(in + rb + cbase + khalf*8);  // global_load_b128
      v4u z = {};
      u = valid ? u : z;                        // 4x v_cndmask_b32
      union { v4u u; h16 h[8]; } cv; cv.u = u;
      #pragma unroll
      for (int j = 0; j < 8; ++j) a[g*8 + j] = cv.h[j];
    } else {                                    // cbase >= CIN (CIN mult of 16)
      #pragma unroll
      for (int j = 0; j < 8; ++j) a[g*8 + j] = (h16)0.0f;
    }
  }
  return a;
}

// ---------------- implicit-GEMM conv3d + fused BN+ReLU ----------------
template<int CIN, int KS>
__global__ __launch_bounds__(256) void conv3d_wmma_kernel(
    const h16* __restrict__ in, const h16* __restrict__ wB,
    const float* __restrict__ gamma, const float* __restrict__ beta,
    h16* __restrict__ out16, float* __restrict__ out32,
    int D, int H, int W, int inD, int inH, int inW,
    int Cout, int Cstride, int coff, int stride, int pad, int mt32)
{
  constexpr int KCH  = (CIN + 31) / 32;
  constexpr int TAPS = KS * KS * KS;
  extern __shared__ char smem_raw[];
  h16* sB = (h16*)smem_raw;
  int tid = threadIdx.x;
  int nt = blockIdx.y;

  // Stage this ntile's full weight block into LDS (global b128 -> ds b128).
  {
    const v8h* gB = (const v8h*)(wB + (long)nt * TAPS * KCH * 512);
    v8h* lB = (v8h*)sB;
    #pragma unroll 1
    for (int i = tid; i < TAPS * KCH * 64; i += 256) lB[i] = gB[i];
  }
  __syncthreads();

  int wave = tid >> 5;
  int lane = tid & 31;
  int m32 = blockIdx.x * 8 + wave;
  if (m32 >= mt32) return;                 // uniform per wave, after barrier

  int row = lane & 15;
  int khalf = lane >> 4;
  int V = D * H * W;

  int xx[2], yy[2], zz[2];
  bool rv[2];
  #pragma unroll
  for (int s = 0; s < 2; ++s) {
    int vm = m32 * 32 + s*16 + row;
    rv[s] = vm < V;
    int vc = rv[s] ? vm : V - 1;
    xx[s] = vc % W; int t = vc / W; yy[s] = t % H; zz[s] = t / H;
  }

  v8f acc0 = {}, acc1 = {};
  int tapIdx = 0;
  #pragma unroll 1
  for (int tz = 0; tz < KS; ++tz) {
    #pragma unroll 1
    for (int ty = 0; ty < KS; ++ty) {
      #pragma unroll 1
      for (int tx = 0; tx < KS; ++tx, ++tapIdx) {
        long rb[2]; bool vd[2];
        #pragma unroll
        for (int s = 0; s < 2; ++s) {
          int iz = zz[s] * stride + tz - pad;
          int iy = yy[s] * stride + ty - pad;
          int ix = xx[s] * stride + tx - pad;
          // branchless: unsigned compare covers [0, dim), bitwise AND (no exec branches)
          vd[s] = rv[s] & ((unsigned)iz < (unsigned)inD)
                        & ((unsigned)iy < (unsigned)inH)
                        & ((unsigned)ix < (unsigned)inW);
          int cz = min(max(iz, 0), inD-1);
          int cy = min(max(iy, 0), inH-1);
          int cx = min(max(ix, 0), inW-1);
          rb[s] = ((long)(cz * inH + cy) * inW + cx) * CIN;
        }
        #pragma unroll
        for (int kc = 0; kc < KCH; ++kc) {
          v16h a0 = load_afrag<CIN>(in, rb[0], vd[0], kc, khalf);
          v16h a1 = load_afrag<CIN>(in, rb[1], vd[1], kc, khalf);
          v16h b = *(const v16h*)(sB + ((tapIdx * KCH + kc) * 32 + lane) * 16); // ds_load_b128 x2
          acc0 = __builtin_amdgcn_wmma_f32_16x16x32_f16(
              false, a0, false, b, (short)0, acc0, false, false);
          acc1 = __builtin_amdgcn_wmma_f32_16x16x32_f16(
              false, a1, false, b, (short)0, acc1, false, false);
        }
      }
    }
  }

  int ncol = nt * 16 + (lane & 15);
  if (ncol < Cout) {
    float sc = gamma[ncol] * BN_INV;
    float bi = beta[ncol];
    #pragma unroll
    for (int s = 0; s < 2; ++s) {
      #pragma unroll
      for (int r = 0; r < 8; ++r) {
        int vr = m32 * 32 + s*16 + r + 8 * khalf;
        if (vr < V) {
          float val = (s ? acc1[r] : acc0[r]) * sc + bi;
          val = val > 0.0f ? val : 0.0f;
          if (out32) out32[(long)vr * Cstride + coff + ncol] = val;
          else       out16[(long)vr * Cstride + coff + ncol] = (h16)val;
        }
      }
    }
  }
}

// ---------------- transpose conv k=2 s=2: 8 parity GEMMs + BN+ReLU ----------------
template<int CIN>
__global__ __launch_bounds__(256) void tconv_wmma_kernel(
    const h16* __restrict__ in, const h16* __restrict__ wB,
    const float* __restrict__ gamma, const float* __restrict__ beta,
    h16* __restrict__ out,
    int inD, int inH, int inW, int Cout, int Cstride, int mt32)
{
  constexpr int KCH = (CIN + 31) / 32;
  extern __shared__ char smem_raw2[];
  h16* sB = (h16*)smem_raw2;
  int tid = threadIdx.x;
  int nt = blockIdx.y;
  int tap = blockIdx.z;                    // 0..7 output parity

  {
    const v8h* gB = (const v8h*)(wB + ((long)nt * 8 + tap) * KCH * 512);
    v8h* lB = (v8h*)sB;
    #pragma unroll 1
    for (int i = tid; i < KCH * 64; i += 256) lB[i] = gB[i];
  }
  __syncthreads();

  int wave = tid >> 5;
  int lane = tid & 31;
  int m32 = blockIdx.x * 8 + wave;
  if (m32 >= mt32) return;

  int row = lane & 15;
  int khalf = lane >> 4;
  int V = inD * inH * inW;

  long rb[2]; bool vd[2];
  #pragma unroll
  for (int s = 0; s < 2; ++s) {
    int vm = m32 * 32 + s*16 + row;
    vd[s] = vm < V;
    rb[s] = (long)(vd[s] ? vm : V - 1) * CIN;
  }

  v8f acc0 = {}, acc1 = {};
  #pragma unroll
  for (int kc = 0; kc < KCH; ++kc) {
    v16h a0 = load_afrag<CIN>(in, rb[0], vd[0], kc, khalf);
    v16h a1 = load_afrag<CIN>(in, rb[1], vd[1], kc, khalf);
    v16h b = *(const v16h*)(sB + (kc * 32 + lane) * 16);
    acc0 = __builtin_amdgcn_wmma_f32_16x16x32_f16(
        false, a0, false, b, (short)0, acc0, false, false);
    acc1 = __builtin_amdgcn_wmma_f32_16x16x32_f16(
        false, a1, false, b, (short)0, acc1, false, false);
  }

  int ncol = nt * 16 + (lane & 15);
  if (ncol < Cout) {
    float sc = gamma[ncol] * BN_INV;
    float bi = beta[ncol];
    int tz = tap >> 2, ty = (tap >> 1) & 1, tx = tap & 1;
    int oH = inH * 2, oW = inW * 2;
    #pragma unroll
    for (int s = 0; s < 2; ++s) {
      #pragma unroll
      for (int r = 0; r < 8; ++r) {
        int vr = m32 * 32 + s*16 + r + 8 * khalf;
        if (vr < V) {
          int ix = vr % inW; int t = vr / inW; int iy = t % inH; int iz = t / inH;
          long o = ((long)(2*iz + tz) * oH + (2*iy + ty)) * oW + (2*ix + tx);
          float val = (s ? acc1[r] : acc0[r]) * sc + bi;
          val = val > 0.0f ? val : 0.0f;
          out[o * Cstride + ncol] = (h16)val;
        }
      }
    }
  }
}

// ---------------- channel-strided copy (skip concat / skip save) ----------------
__global__ __launch_bounds__(256) void copy_chan_kernel(
    const h16* __restrict__ src, h16* __restrict__ dst,
    int Csk, int Ctot, int coff, int total)
{
  int i = blockIdx.x * 256 + threadIdx.x;
  if (i >= total) return;
  int v = i / Csk, c = i - v * Csk;
  dst[(long)v * Ctot + coff + c] = src[i];
}

// ---------------- trilinear devoxelize + L2 normalize ----------------
__global__ __launch_bounds__(256) void devox_kernel(
    const float* __restrict__ vox, const float* __restrict__ coords,
    float* __restrict__ out, int N)
{
  int p = blockIdx.x * 256 + threadIdx.x;
  if (p >= N) return;
  float c0 = coords[p*3+0] * INV_VSIZE - 0.5f;
  float c1 = coords[p*3+1] * INV_VSIZE - 0.5f;
  float c2 = coords[p*3+2] * INV_VSIZE - 0.5f;
  float f0 = floorf(c0), f1 = floorf(c1), f2 = floorf(c2);
  float r0 = c0 - f0, r1 = c1 - f1, r2 = c2 - f2;
  int i0 = (int)f0, i1 = (int)f1, i2 = (int)f2;
  float o[48];
  #pragma unroll
  for (int ch = 0; ch < 48; ++ch) o[ch] = 0.0f;
  #pragma unroll
  for (int dz = 0; dz < 2; ++dz)
    for (int dy = 0; dy < 2; ++dy)
      for (int dx = 0; dx < 2; ++dx) {
        int z = min(max(i0 + dz, 0), GRID-1);
        int y = min(max(i1 + dy, 0), GRID-1);
        int x = min(max(i2 + dx, 0), GRID-1);
        float w = (dz ? r0 : 1.0f - r0) * (dy ? r1 : 1.0f - r1) * (dx ? r2 : 1.0f - r2);
        const float* base = vox + (long)((z * GRID + y) * GRID + x) * 48;
        #pragma unroll
        for (int ch = 0; ch < 48; ++ch) o[ch] += w * base[ch];
      }
  float s = 0.0f;
  #pragma unroll
  for (int ch = 0; ch < 48; ++ch) s += o[ch] * o[ch];
  float nrm = fmaxf(sqrtf(s), 1e-12f);
  float inv = 1.0f / nrm;
  #pragma unroll
  for (int ch = 0; ch < 48; ++ch) out[(long)p * 48 + ch] = o[ch] * inv;
}

// ---------------- host-side templated launch helpers ----------------
template<int CIN, int KS>
static inline void launch_conv(void* const* d_in, const h16* in, h16* out16,
                               float* out32, const h16* wB, int gi, int bi,
                               int D, int inDim, int Cout, int Cstride, int coff,
                               int stride, int pad, hipStream_t stream)
{
  constexpr int KCH = (CIN + 31) / 32, TAPS = KS*KS*KS;
  int nt = (Cout + 15) / 16;
  int V = D * D * D;
  int mt32 = (V + 31) / 32;
  dim3 grid((mt32 + 7) / 8, nt);
  size_t smem = (size_t)TAPS * KCH * 512 * sizeof(h16);
  conv3d_wmma_kernel<CIN, KS><<<grid, 256, smem, stream>>>(
      in, wB, (const float*)d_in[gi], (const float*)d_in[bi],
      out16, out32, D, D, D, inDim, inDim, inDim,
      Cout, Cstride, coff, stride, pad, mt32);
}

template<int CIN>
static inline void launch_tconv(void* const* d_in, const h16* in, h16* out,
                                const h16* wB, int gi, int bi, int inDim,
                                int Cout, int Cstride, hipStream_t stream)
{
  constexpr int KCH = (CIN + 31) / 32;
  int nt = (Cout + 15) / 16;
  int V = inDim * inDim * inDim;
  int mt32 = (V + 31) / 32;
  dim3 grid((mt32 + 7) / 8, nt, 8);
  size_t smem = (size_t)KCH * 512 * sizeof(h16);
  tconv_wmma_kernel<CIN><<<grid, 256, smem, stream>>>(
      in, wB, (const float*)d_in[gi], (const float*)d_in[bi], out,
      inDim, inDim, inDim, Cout, Cstride, mt32);
}

// ---------------- host orchestration ----------------
extern "C" void kernel_launch(void* const* d_in, const int* in_sizes, int n_in,
                              void* d_out, int out_size, void* d_ws, size_t ws_size,
                              hipStream_t stream)
{
  (void)in_sizes; (void)n_in; (void)out_size; (void)ws_size;
  const int V0 = GRID*GRID*GRID;     // 262144
  const int N  = 250000;

  // params flattened after feats,coords,image,py,px (dict keys sorted):
  // down1@5, down2@14, down3@23, down4@32, stem@41, up1@47, up2@56, up3@65, up4@74
  const float* feats  = (const float*)d_in[0];
  const float* coords = (const float*)d_in[1];

  char* ws = (char*)d_ws;
  auto alloc = [&](size_t bytes) -> char* {
    char* p = ws; ws += (bytes + 255) & ~(size_t)255; return p;
  };
  float* gsum  = (float*)alloc((size_t)V0 * 4 * sizeof(float));
  float* gcnt  = (float*)alloc((size_t)V0 * sizeof(float));
  h16*   g16   = (h16*)  alloc((size_t)V0 * 4 * 2);
  h16*   bufA  = (h16*)  alloc((size_t)V0 * 64 * 2);
  h16*   bufB  = (h16*)  alloc((size_t)V0 * 64 * 2);
  h16*   skip0 = (h16*)  alloc((size_t)V0 * 16 * 2);
  h16*   skip1 = (h16*)  alloc((size_t)(V0/8) * 16 * 2);
  h16*   skip2 = (h16*)  alloc((size_t)(V0/64) * 32 * 2);
  h16*   skip3 = (h16*)  alloc((size_t)(V0/512) * 64 * 2);
  float* outv  = (float*)alloc((size_t)V0 * 48 * sizeof(float));
  h16*   wpool = (h16*)  alloc((size_t)12 << 20);
  size_t wcur = 0;

  auto retile = [&](int widx, int Cin, int Cout, int ks, int mirror) -> h16* {
    int taps = ks*ks*ks;
    int kch = (Cin + 31) / 32, nt = (Cout + 15) / 16;
    int total = taps * kch * nt * 512;
    h16* dst = wpool + wcur; wcur += (size_t)total;
    retile_w_kernel<<<(total + 255) / 256, 256, 0, stream>>>(
        (const float*)d_in[widx], dst, Cin, Cout, taps, kch, nt, mirror, total);
    return dst;
  };
  auto copy = [&](const h16* src, h16* dst, int Csk, int Ctot, int coff, int V) {
    int total = V * Csk;
    copy_chan_kernel<<<(total + 255) / 256, 256, 0, stream>>>(src, dst, Csk, Ctot, coff, total);
  };

  // ---- voxelize ----
  hipMemsetAsync(gsum, 0, (size_t)V0 * 4 * sizeof(float), stream);
  hipMemsetAsync(gcnt, 0, (size_t)V0 * sizeof(float), stream);
  vox_scatter_kernel<<<(N + 255) / 256, 256, 0, stream>>>(feats, coords, gsum, gcnt, N);
  vox_final_kernel<<<(V0 + 255) / 256, 256, 0, stream>>>(gsum, gcnt, g16, V0);

  // ---- re-tile all weights (device-side, cheap) ----
  h16* w_s0  = retile(41, 4,   16,  3, 0);
  h16* w_s1  = retile(44, 16,  16,  3, 0);
  h16* w_d1b = retile(5,  16,  16,  2, 0);
  h16* w_d1a = retile(8,  16,  16,  3, 0);
  h16* w_d1c = retile(11, 16,  16,  3, 0);
  h16* w_d2b = retile(14, 16,  32,  2, 0);
  h16* w_d2a = retile(17, 32,  32,  3, 0);
  h16* w_d2c = retile(20, 32,  32,  3, 0);
  h16* w_d3b = retile(23, 32,  64,  2, 0);
  h16* w_d3a = retile(26, 64,  64,  3, 0);
  h16* w_d3c = retile(29, 64,  64,  3, 0);
  h16* w_d4b = retile(32, 64,  128, 2, 0);
  h16* w_d4a = retile(35, 128, 128, 3, 0);
  h16* w_d4c = retile(38, 128, 128, 3, 0);
  h16* w_u1t = retile(47, 128, 128, 2, 1);
  h16* w_u1a = retile(50, 192, 128, 3, 0);
  h16* w_u1c = retile(53, 128, 128, 3, 0);
  h16* w_u2t = retile(56, 128, 64,  2, 1);
  h16* w_u2a = retile(59, 96,  64,  3, 0);
  h16* w_u2c = retile(62, 64,  64,  3, 0);
  h16* w_u3t = retile(65, 64,  48,  2, 1);
  h16* w_u3a = retile(68, 64,  48,  3, 0);
  h16* w_u3c = retile(71, 48,  48,  3, 0);
  h16* w_u4t = retile(74, 48,  48,  2, 1);
  h16* w_u4a = retile(77, 64,  48,  3, 0);
  h16* w_u4c = retile(80, 48,  48,  3, 0);

  // ---- stem (64^3) ----
  launch_conv<4,3>  (d_in, g16,  bufA, nullptr, w_s0, 42, 43, 64, 64, 16, 16, 0, 1, 1, stream);
  launch_conv<16,3> (d_in, bufA, bufB, nullptr, w_s1, 45, 46, 64, 64, 16, 16, 0, 1, 1, stream);
  copy(bufB, skip0, 16, 16, 0, V0);
  // ---- down1 (32^3) ----
  launch_conv<16,2> (d_in, bufB, bufA, nullptr, w_d1b, 6, 7,   32, 64, 16, 16, 0, 2, 0, stream);
  launch_conv<16,3> (d_in, bufA, bufB, nullptr, w_d1a, 9, 10,  32, 32, 16, 16, 0, 1, 1, stream);
  launch_conv<16,3> (d_in, bufB, bufA, nullptr, w_d1c, 12, 13, 32, 32, 16, 16, 0, 1, 1, stream);
  copy(bufA, skip1, 16, 16, 0, V0/8);
  // ---- down2 (16^3) ----
  launch_conv<16,2> (d_in, bufA, bufB, nullptr, w_d2b, 15, 16, 16, 32, 32, 32, 0, 2, 0, stream);
  launch_conv<32,3> (d_in, bufB, bufA, nullptr, w_d2a, 18, 19, 16, 16, 32, 32, 0, 1, 1, stream);
  launch_conv<32,3> (d_in, bufA, bufB, nullptr, w_d2c, 21, 22, 16, 16, 32, 32, 0, 1, 1, stream);
  copy(bufB, skip2, 32, 32, 0, V0/64);
  // ---- down3 (8^3) ----
  launch_conv<32,2> (d_in, bufB, bufA, nullptr, w_d3b, 24, 25, 8, 16, 64, 64, 0, 2, 0, stream);
  launch_conv<64,3> (d_in, bufA, bufB, nullptr, w_d3a, 27, 28, 8, 8,  64, 64, 0, 1, 1, stream);
  launch_conv<64,3> (d_in, bufB, bufA, nullptr, w_d3c, 30, 31, 8, 8,  64, 64, 0, 1, 1, stream);
  copy(bufA, skip3, 64, 64, 0, V0/512);
  // ---- down4 (4^3) ----
  launch_conv<64,2> (d_in, bufA, bufB, nullptr, w_d4b, 33, 34, 4, 8, 128, 128, 0, 2, 0, stream);
  launch_conv<128,3>(d_in, bufB, bufA, nullptr, w_d4a, 36, 37, 4, 4, 128, 128, 0, 1, 1, stream);
  launch_conv<128,3>(d_in, bufA, bufB, nullptr, w_d4c, 39, 40, 4, 4, 128, 128, 0, 1, 1, stream);
  // ---- up1 (8^3), concat 128+64 ----
  launch_tconv<128> (d_in, bufB, bufA, w_u1t, 48, 49, 4, 128, 192, stream);
  copy(skip3, bufA, 64, 192, 128, V0/512);
  launch_conv<192,3>(d_in, bufA, bufB, nullptr, w_u1a, 51, 52, 8, 8, 128, 128, 0, 1, 1, stream);
  launch_conv<128,3>(d_in, bufB, bufA, nullptr, w_u1c, 54, 55, 8, 8, 128, 128, 0, 1, 1, stream);
  // ---- up2 (16^3), concat 64+32 ----
  launch_tconv<128> (d_in, bufA, bufB, w_u2t, 57, 58, 8, 64, 96, stream);
  copy(skip2, bufB, 32, 96, 64, V0/64);
  launch_conv<96,3> (d_in, bufB, bufA, nullptr, w_u2a, 60, 61, 16, 16, 64, 64, 0, 1, 1, stream);
  launch_conv<64,3> (d_in, bufA, bufB, nullptr, w_u2c, 63, 64, 16, 16, 64, 64, 0, 1, 1, stream);
  // ---- up3 (32^3), concat 48+16 ----
  launch_tconv<64>  (d_in, bufB, bufA, w_u3t, 66, 67, 16, 48, 64, stream);
  copy(skip1, bufA, 16, 64, 48, V0/8);
  launch_conv<64,3> (d_in, bufA, bufB, nullptr, w_u3a, 69, 70, 32, 32, 48, 48, 0, 1, 1, stream);
  launch_conv<48,3> (d_in, bufB, bufA, nullptr, w_u3c, 72, 73, 32, 32, 48, 48, 0, 1, 1, stream);
  // ---- up4 (64^3), concat 48+16 ----
  launch_tconv<48>  (d_in, bufA, bufB, w_u4t, 75, 76, 32, 48, 64, stream);
  copy(skip0, bufB, 16, 64, 48, V0);
  launch_conv<64,3> (d_in, bufB, bufA, nullptr, w_u4a, 78, 79, 64, 64, 48, 48, 0, 1, 1, stream);
  launch_conv<48,3> (d_in, bufA, nullptr, outv, w_u4c, 81, 82, 64, 64, 48, 48, 0, 1, 1, stream);

  // ---- devoxelize + normalize ----
  devox_kernel<<<(N + 255) / 256, 256, 0, stream>>>(outv, coords, (float*)d_out, N);
}
